// Attention_9371618640426
// MI455X (gfx1250) — matrix operand order repaired and checked by reading, not compiled, not run on hardware
//
#include <hip/hip_runtime.h>

#define DEV __device__ __forceinline__

typedef __attribute__((ext_vector_type(16))) _Float16 v16h;
typedef __attribute__((ext_vector_type(8)))  _Float16 v8h;
typedef __attribute__((ext_vector_type(8)))  float    v8f;
typedef __attribute__((ext_vector_type(4)))  unsigned int v4u;
typedef __attribute__((ext_vector_type(8)))  unsigned int v8u;

// Problem constants (fixed by the reference)
constexpr int Bn = 4, S = 2048, D = 1024, H = 16, HD = 64;
constexpr int M = Bn * S;   // 8192
constexpr int N = D;        // 1024
constexpr int Kd = D;       // 1024

DEV v8f wmma16(v16h a, v16h b, v8f c) {
  return __builtin_amdgcn_wmma_f32_16x16x32_f16(
      false, a, false, b, (short)0, c, false, false);
}

// A-fragment (16x32 f16, MxK) from row-major [rows][stride] LDS tile.
DEV v16h ldsA(const _Float16* base, int stride, int row0, int k0) {
  const int lane = threadIdx.x & 31;
  const _Float16* p = base + (size_t)(row0 + (lane & 15)) * stride + k0 + ((lane >> 4) << 3);
  v16h f;
#pragma unroll
  for (int i = 0; i < 8; ++i) { f[i] = p[i]; f[i + 8] = p[i + 16]; }
  return f;
}

// B-fragment (32x16 f16, KxN) from an [N][K] row-major LDS tile.
DEV v16h ldsB(const _Float16* base, int stride, int row0, int k0) {
  const int lane = threadIdx.x & 31;
  const _Float16* p = base + (size_t)(row0 + (lane & 15)) * stride + k0 + ((lane >> 4) << 3);
  v16h f;
#pragma unroll
  for (int i = 0; i < 8; ++i) { f[i] = p[i]; f[i + 8] = p[i + 16]; }
  return f;
}

// ---------------------------------------------------------------------------
// Tensor Data Mover: 2-D tile load Global->LDS with row padding.
// Tile: tile_d1 rows x tile_d0 elems (f16). LDS rows padded +16B (stride 72h).
// pad_interval=4 (pad after 32 DWORDs = 128B = one 64-half row),
// pad_amount=3 (4 DWORDs = 16B).
// ---------------------------------------------------------------------------
DEV void tdm_load_2d_f16(unsigned lds_off, const _Float16* gptr,
                         unsigned tensor_d0, unsigned tensor_d1,
                         unsigned stride0) {
  const unsigned long long ga = (unsigned long long)(uintptr_t)gptr;
  v4u g0;
  g0[0] = 1u;                                   // count=1, user mode
  g0[1] = lds_off;                              // lds_addr [63:32]
  g0[2] = (unsigned)ga;                         // global_addr [95:64]
  g0[3] = (unsigned)(ga >> 32) | 0x80000000u;   // global_addr hi | type=2
  v8u g1;
  // wg_mask=0 | data_size=1(2B) | pad_enable | pad_interval=4 | pad_amount=3
  g1[0] = 0x00010000u | 0x00100000u | (4u << 22) | (3u << 25);
  g1[1] = (tensor_d0 & 0xFFFFu) << 16;                       // tensor_dim0 lo
  g1[2] = (tensor_d0 >> 16) | ((tensor_d1 & 0xFFFFu) << 16); // d0 hi | d1 lo
  g1[3] = (tensor_d1 >> 16) | (64u << 16);                   // d1 hi | tile_dim0=64
  g1[4] = 64u;                                               // tile_dim1=64, tile_dim2=0
  g1[5] = stride0;                                           // tensor_dim0_stride lo
  g1[6] = 0u;
  g1[7] = 0u;
  asm volatile("tensor_load_to_lds %0, %1" :: "s"(g0), "s"(g1) : "memory");
}

// ---------------------------------------------------------------------------
// GEMM: out = A[M,K] * W[K,N] + bias, tile 128x128, 8 waves (4x2), wave=32x64.
// MODE 0: f16 out -> [B,H,S,HD] (Q,K)   MODE 1: f16 out -> [B,H,HD,S] (V)
// MODE 2: f32 out row-major [M,N]
// ---------------------------------------------------------------------------
template <int MODE, typename TIN>
__global__ __launch_bounds__(256) void gemm_kernel(
    const TIN* __restrict__ A, const float* __restrict__ W,
    const float* __restrict__ bias, void* __restrict__ outp) {
  __shared__ _Float16 a_s[128 * 40];  // [m][k] padded
  __shared__ _Float16 b_s[128 * 40];  // [n][k] (W transposed) padded

  const int tid = threadIdx.x;
  const int m0 = blockIdx.y * 128;
  const int n0 = blockIdx.x * 128;
  const int w = tid >> 5, lane = tid & 31;
  const int wm = (w >> 1) * 32, wn = (w & 1) * 64;

  v8f acc[2][4];
#pragma unroll
  for (int mf = 0; mf < 2; ++mf)
#pragma unroll
    for (int nf = 0; nf < 4; ++nf) acc[mf][nf] = {};

  for (int kt = 0; kt < Kd; kt += 32) {
    __syncthreads();
    // A tile 128x32: one b128 LDS store of 8 converted halves per chunk
#pragma unroll
    for (int i = 0; i < 2; ++i) {
      const int ch = tid + i * 256;            // 0..511
      const int r = ch >> 2, c8 = (ch & 3) * 8;
      const TIN* src = &A[(size_t)(m0 + r) * Kd + kt + c8];
      v8h hv;
#pragma unroll
      for (int j = 0; j < 8; ++j) hv[j] = (_Float16)src[j];
      *(v8h*)&a_s[r * 40 + c8] = hv;
    }
    // W tile 32x128 -> [n][k]: gather 8 k per thread (coalesced across lanes)
#pragma unroll
    for (int i = 0; i < 2; ++i) {
      const int ch = tid + i * 256;
      const int nn = ch & 127, k8 = (ch >> 7) * 8;
      v8h hv;
#pragma unroll
      for (int j = 0; j < 8; ++j)
        hv[j] = (_Float16)W[(size_t)(kt + k8 + j) * N + n0 + nn];
      *(v8h*)&b_s[nn * 40 + k8] = hv;
    }
    __syncthreads();

    v16h af[2];
#pragma unroll
    for (int mf = 0; mf < 2; ++mf) af[mf] = ldsA(a_s, 40, wm + mf * 16, 0);
#pragma unroll
    for (int nf = 0; nf < 4; ++nf) {
      const v16h bf = ldsB(b_s, 40, wn + nf * 16, 0);
#pragma unroll
      for (int mf = 0; mf < 2; ++mf) acc[mf][nf] = wmma16(af[mf], bf, acc[mf][nf]);
    }
  }

  const int mrow = (lane >> 4) << 3;
#pragma unroll
  for (int mf = 0; mf < 2; ++mf)
#pragma unroll
    for (int nf = 0; nf < 4; ++nf) {
      const int n = n0 + wn + nf * 16 + (lane & 15);
      const float bv = bias[n];
      if (MODE == 1) {
        // [B,H,HD,S]: 8 consecutive s per lane -> one b128 store
        const int mb = m0 + wm + mf * 16 + mrow;
        const int b_ = mb >> 11, s = mb & (S - 1);
        const int h = n >> 6, d = n & (HD - 1);
        v8h hv;
#pragma unroll
        for (int r = 0; r < 8; ++r) hv[r] = (_Float16)(acc[mf][nf][r] + bv);
        *(v8h*)&((_Float16*)outp)[(((size_t)b_ * H + h) * HD + d) * S + s] = hv;
      } else {
#pragma unroll
        for (int r = 0; r < 8; ++r) {
          const int m = m0 + wm + mf * 16 + r + mrow;
          const float v = acc[mf][nf][r] + bv;
          if (MODE == 0) {
            const int b_ = m >> 11, s = m & (S - 1), h = n >> 6, d = n & (HD - 1);
            ((_Float16*)outp)[(((size_t)b_ * H + h) * S + s) * HD + d] = (_Float16)v;
          } else {
            ((float*)outp)[(size_t)m * N + n] = v;
          }
        }
      }
    }
}

// ---------------------------------------------------------------------------
// Flash attention, causal. Grid: (S/64, B*H). Block: 128 threads (4 waves).
// K/V tiles staged by the Tensor Data Mover; online softmax in C-fragment
// layout; P re-enters A-layout via per-wave LDS rows.
// ---------------------------------------------------------------------------
__global__ __launch_bounds__(128) void flash_kernel(
    const _Float16* __restrict__ qg, const _Float16* __restrict__ kg,
    const _Float16* __restrict__ vtg, _Float16* __restrict__ og) {
  __shared__ _Float16 q_s[64 * 72];
  __shared__ _Float16 k_s[64 * 72];   // [key][d]
  __shared__ _Float16 vt_s[64 * 72];  // [d][key]
  __shared__ _Float16 p_s[64 * 72];   // per-wave rows; reused for O epilogue

  const int tid = threadIdx.x;
  const int bh = blockIdx.y;
  const int q0 = blockIdx.x * 64;
  const _Float16* Q = qg + (size_t)bh * S * HD;
  const _Float16* Kp = kg + (size_t)bh * S * HD;
  const _Float16* Vt = vtg + (size_t)bh * HD * S;

#pragma unroll
  for (int i = 0; i < 4; ++i) {
    const int ch = tid + i * 128;               // 0..511 chunks of 8 halves
    const int r = ch >> 3, c8 = (ch & 7) * 8;
    *(v8h*)&q_s[r * 72 + c8] = *(const v8h*)&Q[(size_t)(q0 + r) * HD + c8];
  }

  const int w = tid >> 5, lane = tid & 31;
  const int mrow = (lane >> 4) << 3;

  float mi[8], li[8];
  v8f oacc[4];
#pragma unroll
  for (int r = 0; r < 8; ++r) { mi[r] = -1e30f; li[r] = 0.f; }
#pragma unroll
  for (int nf = 0; nf < 4; ++nf) oacc[nf] = {};

  const int ntiles = blockIdx.x + 1;
  for (int t = 0; t < ntiles; ++t) {
    const int k0 = t * 64;
    __syncthreads();  // previous-tile consumers done before TDM overwrites
    if (w == 0) {     // K tile: rows k0..k0+63 of [S][64]
      tdm_load_2d_f16((unsigned)(uintptr_t)k_s, Kp + (size_t)k0 * HD,
                      64u, (unsigned)S, 64u);
      __builtin_amdgcn_s_wait_tensorcnt(0);
    } else if (w == 1) {  // Vt tile: 64 rows x cols k0..k0+63 of [64][S]
      tdm_load_2d_f16((unsigned)(uintptr_t)vt_s, Vt + k0,
                      (unsigned)S, 64u, (unsigned)S);
      __builtin_amdgcn_s_wait_tensorcnt(0);
    }
    __syncthreads();

    // S = Q * K^T
    v8f sacc[4];
#pragma unroll
    for (int nf = 0; nf < 4; ++nf) sacc[nf] = {};
#pragma unroll
    for (int kk = 0; kk < HD; kk += 32) {
      const v16h a = ldsA(q_s, 72, w * 16, kk);
#pragma unroll
      for (int nf = 0; nf < 4; ++nf) {
        const v16h b = ldsB(k_s, 72, nf * 16, kk);
        sacc[nf] = wmma16(a, b, sacc[nf]);
      }
    }

    // scale + causal mask + tile row-max
    float tmax[8];
#pragma unroll
    for (int r = 0; r < 8; ++r) tmax[r] = -1e30f;
#pragma unroll
    for (int nf = 0; nf < 4; ++nf)
#pragma unroll
      for (int r = 0; r < 8; ++r) {
        float sv = sacc[nf][r] * 0.125f;
        const int col = k0 + nf * 16 + (lane & 15);
        const int row = q0 + w * 16 + r + mrow;
        sv = (col > row) ? -1e30f : sv;
        sacc[nf][r] = sv;
        tmax[r] = fmaxf(tmax[r], sv);
      }
#pragma unroll
    for (int r = 0; r < 8; ++r) {
      tmax[r] = fmaxf(tmax[r], __shfl_xor(tmax[r], 1, 32));
      tmax[r] = fmaxf(tmax[r], __shfl_xor(tmax[r], 2, 32));
      tmax[r] = fmaxf(tmax[r], __shfl_xor(tmax[r], 4, 32));
      tmax[r] = fmaxf(tmax[r], __shfl_xor(tmax[r], 8, 32));
    }

    float corr[8], rs[8];
#pragma unroll
    for (int r = 0; r < 8; ++r) {
      const float mn = fmaxf(mi[r], tmax[r]);
      corr[r] = __expf(mi[r] - mn);
      mi[r] = mn;
      rs[r] = 0.f;
    }
#pragma unroll
    for (int nf = 0; nf < 4; ++nf)
#pragma unroll
      for (int r = 0; r < 8; ++r) {
        const float p = __expf(sacc[nf][r] - mi[r]);
        sacc[nf][r] = p;
        rs[r] += p;
      }
#pragma unroll
    for (int r = 0; r < 8; ++r) {
      rs[r] += __shfl_xor(rs[r], 1, 32);
      rs[r] += __shfl_xor(rs[r], 2, 32);
      rs[r] += __shfl_xor(rs[r], 4, 32);
      rs[r] += __shfl_xor(rs[r], 8, 32);
      li[r] = li[r] * corr[r] + rs[r];
    }
#pragma unroll
    for (int nf = 0; nf < 4; ++nf)
#pragma unroll
      for (int r = 0; r < 8; ++r) oacc[nf][r] *= corr[r];

    // P -> per-wave LDS rows -> A-fragments
#pragma unroll
    for (int nf = 0; nf < 4; ++nf)
#pragma unroll
      for (int r = 0; r < 8; ++r)
        p_s[(size_t)(w * 16 + r + mrow) * 72 + nf * 16 + (lane & 15)] =
            (_Float16)sacc[nf][r];

    // O += P * V
#pragma unroll
    for (int kk = 0; kk < 64; kk += 32) {
      const v16h a = ldsA(p_s, 72, w * 16, kk);
#pragma unroll
      for (int nf = 0; nf < 4; ++nf) {
        const v16h b = ldsB(vt_s, 72, nf * 16, kk);
        oacc[nf] = wmma16(a, b, oacc[nf]);
      }
    }
  }

  // Normalize, stage tile in LDS, then coalesced b128 stores to [B,S,D].
  float inv[8];
#pragma unroll
  for (int r = 0; r < 8; ++r) inv[r] = 1.0f / li[r];
#pragma unroll
  for (int nf = 0; nf < 4; ++nf)
#pragma unroll
    for (int r = 0; r < 8; ++r)
      p_s[(size_t)(w * 16 + r + mrow) * 72 + nf * 16 + (lane & 15)] =
          (_Float16)(oacc[nf][r] * inv[r]);
  __syncthreads();

  const int b_ = bh >> 4, h = bh & 15;
#pragma unroll
  for (int i = 0; i < 4; ++i) {
    const int ch = tid + i * 128;
    const int r = ch >> 3, c8 = (ch & 7) * 8;
    *(v8h*)&og[((size_t)b_ * S + q0 + r) * D + h * HD + c8] =
        *(const v8h*)&p_s[r * 72 + c8];
  }
}

// ---------------------------------------------------------------------------
extern "C" void kernel_launch(void* const* d_in, const int* in_sizes, int n_in,
                              void* d_out, int out_size, void* d_ws, size_t ws_size,
                              hipStream_t stream) {
  const float* query = (const float*)d_in[0];
  const float* key   = (const float*)d_in[1];
  const float* value = (const float*)d_in[2];
  const float* Wq = (const float*)d_in[3];
  const float* bq = (const float*)d_in[4];
  const float* Wk = (const float*)d_in[5];
  const float* bk = (const float*)d_in[6];
  const float* Wv = (const float*)d_in[7];
  const float* bv = (const float*)d_in[8];
  const float* Wo = (const float*)d_in[9];
  const float* bo = (const float*)d_in[10];

  const size_t per = (size_t)Bn * S * D;
  _Float16* q16 = (_Float16*)d_ws;  // [B,H,S,HD]
  _Float16* k16 = q16 + per;        // [B,H,S,HD]
  _Float16* v16 = k16 + per;        // [B,H,HD,S]
  _Float16* o16 = v16 + per;        // [B,S,D]

  dim3 gg(N / 128, M / 128);
  gemm_kernel<0, float><<<gg, 256, 0, stream>>>(query, Wq, bq, (void*)q16);
  gemm_kernel<0, float><<<gg, 256, 0, stream>>>(key, Wk, bk, (void*)k16);
  gemm_kernel<1, float><<<gg, 256, 0, stream>>>(value, Wv, bv, (void*)v16);

  dim3 fg(S / 64, Bn * H);
  flash_kernel<<<fg, 128, 0, stream>>>(q16, k16, v16, o16);

  gemm_kernel<2, _Float16><<<gg, 256, 0, stream>>>(o16, Wo, bo, d_out);
}
